// EI_Adaptive_Fusion_2671469658235
// MI455X (gfx1250) — matrix operand order repaired and checked by reading, not compiled, not run on hardware
//
#include <hip/hip_runtime.h>
#include <hip/hip_bf16.h>
#include <math.h>

typedef __bf16 bf16_t;
typedef __attribute__((ext_vector_type(16))) __bf16 v16bf;
typedef __attribute__((ext_vector_type(8)))  __bf16 v8bf;
typedef __attribute__((ext_vector_type(8)))  float   v8f;

#define B_   4
#define C_   128
#define NQ   512
#define NKV  642
#define NKP  672   // padded kv count (multiple of 32)
#define PI_F 3.14159265358979323846f

// ---------------- WMMA tile helpers (CDNA5 16x16x32 bf16 layouts) ----------------
// A (16x32, MxK): lane l holds row M=l&15; element j <-> K = j + 8*(l>>4) + 8*(j>=8)
__device__ __forceinline__ v16bf load_a_tile(const bf16_t* __restrict__ A, int lda,
                                             int mbase, int k0, int lane) {
  int g = lane >> 4;
  const bf16_t* p = A + (size_t)(mbase + (lane & 15)) * lda + k0 + 8 * g;
  union { v16bf v; v8bf h[2]; } u;
  u.h[0] = *(const v8bf*)p;
  u.h[1] = *(const v8bf*)(p + 16);
  return u.v;
}
// B (32x16, KxN): lane l holds column N=l&15; element j <-> K = j + 8*(l>>4) + 8*(j>=8)
__device__ __forceinline__ v16bf load_b_tile(const bf16_t* __restrict__ Bm, int ldb,
                                             int k0, int nbase, int lane) {
  int g = lane >> 4;
  const bf16_t* p = Bm + (size_t)(k0 + 8 * g) * ldb + nbase + (lane & 15);
  v16bf v;
#pragma unroll
  for (int j = 0; j < 16; ++j) v[j] = p[(size_t)(j + 8 * (j >= 8)) * ldb];
  return v;
}
__device__ __forceinline__ v8f wmma_bf16(v16bf a, v16bf b, v8f c) {
  return __builtin_amdgcn_wmma_f32_16x16x32_bf16(false, a, false, b, (short)0, c, false, false);
}

// ---------------- prep: transposes, bf16 casts, padding, coords, Wdelta row-sums ----
__global__ void k_prep(const float* __restrict__ erp_f, const float* __restrict__ ico,
                       const float* __restrict__ Wq_a, const float* __restrict__ Wq_d,
                       const float* __restrict__ Wkv_a, const float* __restrict__ Wkv_d,
                       const float* __restrict__ Wp_a, const float* __restrict__ Wp_d,
                       const float* __restrict__ Wga, const float* __restrict__ Wgd,
                       const float* __restrict__ Wdelta,
                       bf16_t* __restrict__ erp_bf, bf16_t* __restrict__ ico_bf,
                       bf16_t* __restrict__ Wq_cat, bf16_t* __restrict__ Wkv_cat,
                       bf16_t* __restrict__ Wpa_bf, bf16_t* __restrict__ Wpd_bf,
                       bf16_t* __restrict__ Wg_cat, float* __restrict__ qcoord,
                       float* __restrict__ wsum) {
  const int n0 = B_ * NQ * C_;
  const int n1 = n0 + B_ * NKP * C_;
  const int n2 = n1 + 128 * 256;
  const int n3 = n2 + 128 * 512;
  const int n4 = n3 + 128 * 128;
  const int n5 = n4 + 128 * 128;
  const int n6 = n5 + 256 * 256;
  const int n7 = n6 + NQ;
  const int n8 = n7 + 3;
  for (int i = blockIdx.x * blockDim.x + threadIdx.x; i < n8; i += gridDim.x * blockDim.x) {
    if (i < n0) {                       // erp (B,C,HW) -> (B,HW,C) bf16
      int b = i / (NQ * C_); int r = i % (NQ * C_); int n = r / C_, c = r % C_;
      erp_bf[i] = (bf16_t)erp_f[((size_t)b * C_ + c) * NQ + n];
    } else if (i < n1) {                // ico zero-padded to 672 rows, bf16
      int j = i - n0; int b = j / (NKP * C_); int r = (j / C_) % NKP; int c = j % C_;
      ico_bf[j] = (r < NKV) ? (bf16_t)ico[((size_t)b * NKV + r) * C_ + c] : (bf16_t)0.0f;
    } else if (i < n2) {                // [Wq_a | Wq_d] (128x256)
      int j = i - n1; int k = j / 256, n = j % 256;
      Wq_cat[j] = (bf16_t)((n < 128) ? Wq_a[k * 128 + n] : Wq_d[k * 128 + (n - 128)]);
    } else if (i < n3) {                // [Wkv_a | Wkv_d] (128x512)
      int j = i - n2; int k = j / 512, n = j % 512;
      Wkv_cat[j] = (bf16_t)((n < 256) ? Wkv_a[k * 256 + n] : Wkv_d[k * 256 + (n - 256)]);
    } else if (i < n4) { Wpa_bf[i - n3] = (bf16_t)Wp_a[i - n3];
    } else if (i < n5) { Wpd_bf[i - n4] = (bf16_t)Wp_d[i - n4];
    } else if (i < n6) {                // [Wga | Wgd] (256x256)
      int j = i - n5; int k = j / 256, n = j % 256;
      Wg_cat[j] = (bf16_t)((n < 128) ? Wga[k * 128 + n] : Wgd[k * 128 + (n - 128)]);
    } else if (i < n7) {                // ERP sphere coords (512x3)
      int n = i - n6; int y = n >> 5, x = n & 31;
      float u = (x - 16.5f) * (2.0f * PI_F / 32.0f);
      float v = (y - 8.5f) * (PI_F / 16.0f);
      float cv = __cosf(v), sv = __sinf(v);
      qcoord[n * 3 + 0] = cv * __sinf(u);
      qcoord[n * 3 + 1] = sv;
      qcoord[n * 3 + 2] = cv * __cosf(u);
    } else {                            // wsum[d] = sum_c Wdelta[d][c]
      int d = i - n7; float s = 0.0f;
      for (int c = 0; c < 128; ++c) s += Wdelta[d * 128 + c];
      wsum[d] = s;
    }
  }
}

// ---------------- Q projection: q_a (scaled bf16) and q_d (f32) -------------------
__global__ __launch_bounds__(32) void k_gemm_q(const bf16_t* __restrict__ erp,
                                               const bf16_t* __restrict__ Wq,
                                               bf16_t* __restrict__ qa,
                                               float* __restrict__ qd) {
  int lane = threadIdx.x, mbase = blockIdx.x * 16, nbase = blockIdx.y * 16, b = blockIdx.z;
  const bf16_t* A = erp + (size_t)b * NQ * C_;
  v8f acc = {};
  for (int k0 = 0; k0 < 128; k0 += 32)
    acc = wmma_bf16(load_a_tile(A, 128, mbase, k0, lane), load_b_tile(Wq, 256, k0, nbase, lane), acc);
  int g = lane >> 4, nl = lane & 15;
  size_t rowbase = (size_t)b * NQ + mbase + 8 * g;
  if (blockIdx.y < 8) {                 // uniform branch hoisted out of store loop
    bf16_t* dst = qa + rowbase * C_ + blockIdx.y * 16 + nl;
#pragma unroll
    for (int r = 0; r < 8; ++r) dst[(size_t)r * C_] = (bf16_t)(acc[r] * 0.08838834764831845f);
  } else {
    float* dst = qd + rowbase * C_ + (blockIdx.y - 8) * 16 + nl;
#pragma unroll
    for (int r = 0; r < 8; ++r) dst[(size_t)r * C_] = acc[r];
  }
}

// ---------------- KV projection: kT_a (bf16, transposed), v_a, k_d (f32), v_d ------
__global__ __launch_bounds__(32) void k_gemm_kv(const bf16_t* __restrict__ ico_bf,
                                                const bf16_t* __restrict__ Wkv,
                                                bf16_t* __restrict__ kTa,
                                                bf16_t* __restrict__ va,
                                                float* __restrict__ kd,
                                                bf16_t* __restrict__ vd) {
  int lane = threadIdx.x, mbase = blockIdx.x * 16, nbase = blockIdx.y * 16, b = blockIdx.z;
  const bf16_t* A = ico_bf + (size_t)b * NKP * C_;
  v8f acc = {};
  for (int k0 = 0; k0 < 128; k0 += 32)
    acc = wmma_bf16(load_a_tile(A, 128, mbase, k0, lane), load_b_tile(Wkv, 512, k0, nbase, lane), acc);
  int g = lane >> 4, nl = lane & 15;
  int sec = blockIdx.y >> 3;
  int col = (blockIdx.y & 7) * 16 + nl;
  size_t rowbase = (size_t)b * NKP + mbase + 8 * g;
  if (sec == 0) {                       // K^T: 8 consecutive m for fixed column -> one 16B store
    union { v8bf v; bf16_t e[8]; } u;
#pragma unroll
    for (int r = 0; r < 8; ++r) u.e[r] = (bf16_t)acc[r];
    *(v8bf*)(kTa + ((size_t)b * C_ + col) * NKP + mbase + 8 * g) = u.v;
  } else if (sec == 1) {
    bf16_t* dst = va + rowbase * C_ + col;
#pragma unroll
    for (int r = 0; r < 8; ++r) dst[(size_t)r * C_] = (bf16_t)acc[r];
  } else if (sec == 2) {
    float* dst = kd + rowbase * C_ + col;
#pragma unroll
    for (int r = 0; r < 8; ++r) dst[(size_t)r * C_] = acc[r];
  } else {
    bf16_t* dst = vd + rowbase * C_ + col;
#pragma unroll
    for (int r = 0; r < 8; ++r) dst[(size_t)r * C_] = (bf16_t)acc[r];
  }
}

// ---------------- generic batched bf16 GEMM -> f32 (logits & attn@V) ---------------
__global__ __launch_bounds__(32) void k_gemm_f32out(const bf16_t* __restrict__ A, int lda, long long sA,
                                                    const bf16_t* __restrict__ Bm, int ldb, long long sB,
                                                    float* __restrict__ Out, int ldo, long long sO, int K) {
  int lane = threadIdx.x, mbase = blockIdx.x * 16, nbase = blockIdx.y * 16, b = blockIdx.z;
  const bf16_t* Ab = A + (size_t)b * sA;
  const bf16_t* Bb = Bm + (size_t)b * sB;
  float* Ob = Out + (size_t)b * sO;
  v8f acc = {};
  for (int k0 = 0; k0 < K; k0 += 32)
    acc = wmma_bf16(load_a_tile(Ab, lda, mbase, k0, lane), load_b_tile(Bb, ldb, k0, nbase, lane), acc);
  int g = lane >> 4, nl = lane & 15;
  float* dst = Ob + (size_t)(mbase + 8 * g) * ldo + nbase + nl;
#pragma unroll
  for (int r = 0; r < 8; ++r) dst[(size_t)r * ldo] = acc[r];
}

// ---------------- difference-branch logits: sum_c exp(-|q-k|) + pos ----------------
#define QB 8
__global__ __launch_bounds__(256) void k_sdiff(const float* __restrict__ qd,
                                               const float* __restrict__ qc,
                                               const float* __restrict__ kdm,
                                               const float* __restrict__ kcoord,
                                               const float* __restrict__ wsum,
                                               float* __restrict__ Sd) {
  int b = blockIdx.y, q0 = blockIdx.x * QB, tid = threadIdx.x;
  __shared__ float qs[QB][C_];
  __shared__ float qcs[QB][3];
  for (int i = tid; i < QB * C_; i += 256)
    qs[i / C_][i % C_] = qd[((size_t)b * NQ + q0 + i / C_) * C_ + (i % C_)];
  for (int i = tid; i < QB * 3; i += 256)
    qcs[i / 3][i % 3] = qc[(q0 + i / 3) * 3 + (i % 3)];
  __syncthreads();
  float w0 = wsum[0], w1 = wsum[1], w2 = wsum[2];
  for (int k = tid; k < NKV; k += 256) {
    const float* kr = kdm + ((size_t)b * NKP + k) * C_;
    if (k + 256 < NKV)                  // gfx1250 global_prefetch of next streamed row
      __builtin_prefetch(kdm + ((size_t)b * NKP + k + 256) * C_, 0, 1);
    float acc[QB];
#pragma unroll
    for (int qi = 0; qi < QB; ++qi) acc[qi] = 0.0f;
    for (int c = 0; c < C_; c += 4) {
      float4 kv = *(const float4*)(kr + c);
#pragma unroll
      for (int qi = 0; qi < QB; ++qi) {
        acc[qi] += __expf(-fabsf(qs[qi][c    ] - kv.x));
        acc[qi] += __expf(-fabsf(qs[qi][c + 1] - kv.y));
        acc[qi] += __expf(-fabsf(qs[qi][c + 2] - kv.z));
        acc[qi] += __expf(-fabsf(qs[qi][c + 3] - kv.w));
      }
    }
    float kc0 = kcoord[((size_t)b * NKV + k) * 3 + 0];
    float kc1 = kcoord[((size_t)b * NKV + k) * 3 + 1];
    float kc2 = kcoord[((size_t)b * NKV + k) * 3 + 2];
#pragma unroll
    for (int qi = 0; qi < QB; ++qi) {
      float pos = w0 * __expf(-fabsf(qcs[qi][0] - kc0))
                + w1 * __expf(-fabsf(qcs[qi][1] - kc1))
                + w2 * __expf(-fabsf(qcs[qi][2] - kc2));
      Sd[((size_t)b * NQ + q0 + qi) * NKP + k] = acc[qi] + pos;
    }
  }
}

// ---------------- masked row softmax (wave per row) -> bf16 probabilities ----------
__global__ __launch_bounds__(32) void k_softmax(const float* __restrict__ S,
                                                bf16_t* __restrict__ P, float scale) {
  int row = blockIdx.x, lane = threadIdx.x;
  const float* s = S + (size_t)row * NKP;
  bf16_t* p = P + (size_t)row * NKP;
  float v[NKP / 32];
#pragma unroll
  for (int j = 0; j < NKP / 32; ++j) {
    int c = lane + j * 32;
    v[j] = (c < NKV) ? s[c] * scale : -3.0e38f;
  }
  float mx = v[0];
#pragma unroll
  for (int j = 1; j < NKP / 32; ++j) mx = fmaxf(mx, v[j]);
  for (int m = 16; m >= 1; m >>= 1) mx = fmaxf(mx, __shfl_xor(mx, m, 32));
  float sum = 0.0f;
#pragma unroll
  for (int j = 0; j < NKP / 32; ++j) { v[j] = __expf(v[j] - mx); sum += v[j]; }
  for (int m = 16; m >= 1; m >>= 1) sum += __shfl_xor(sum, m, 32);
  float inv = 1.0f / sum;
#pragma unroll
  for (int j = 0; j < NKP / 32; ++j)    // masked entries are exactly 0 -> store unconditionally
    p[lane + j * 32] = (bf16_t)(v[j] * inv);
}

// ---------------- reference's transpose/reshape quirk + bf16 cast ------------------
__global__ void k_perm_cvt(const float* __restrict__ Oa, const float* __restrict__ Od,
                           bf16_t* __restrict__ Xa, bf16_t* __restrict__ Xd) {
  int i = blockIdx.x * 256 + threadIdx.x;
  if (i >= B_ * NQ * C_) return;
  int b = i / (NQ * C_);
  int f = i % (NQ * C_);
  Xa[i] = (bf16_t)Oa[((size_t)b * NQ + (f & 511)) * C_ + (f >> 9)];
  Xd[i] = (bf16_t)Od[i];
}

// ---------------- Wp projection + bias; writes f32 result and bf16 cat slice -------
__global__ __launch_bounds__(32) void k_gemm_p(const bf16_t* __restrict__ A,
                                               const bf16_t* __restrict__ W,
                                               const float* __restrict__ bias,
                                               float* __restrict__ outf,
                                               bf16_t* __restrict__ cat, int catOff) {
  int lane = threadIdx.x, mbase = blockIdx.x * 16, nbase = blockIdx.y * 16, b = blockIdx.z;
  const bf16_t* Ab = A + (size_t)b * NQ * C_;
  v8f acc = {};
  for (int k0 = 0; k0 < 128; k0 += 32)
    acc = wmma_bf16(load_a_tile(Ab, 128, mbase, k0, lane), load_b_tile(W, 128, k0, nbase, lane), acc);
  int g = lane >> 4, nl = lane & 15, col = nbase + nl;
  float bv = bias[col];
  size_t rowbase = (size_t)b * NQ + mbase + 8 * g;
  float*  df = outf + rowbase * C_ + col;
  bf16_t* dc = cat + rowbase * 256 + catOff + col;
#pragma unroll
  for (int r = 0; r < 8; ++r) {
    float val = acc[r] + bv;
    df[(size_t)r * C_] = val;
    dc[(size_t)r * 256] = (bf16_t)val;
  }
}

// ---------------- gating GEMM + sigmoid blend + output transpose -------------------
__global__ __launch_bounds__(32) void k_gate_fuse(const bf16_t* __restrict__ cat,
                                                  const bf16_t* __restrict__ Wg,
                                                  const float* __restrict__ xa,
                                                  const float* __restrict__ xd,
                                                  float* __restrict__ out) {
  int lane = threadIdx.x, mbase = blockIdx.x * 16, nbase = blockIdx.y * 16, b = blockIdx.z;
  const bf16_t* Ab = cat + (size_t)b * NQ * 256;
  v8f accA = {}, accD = {};
  for (int k0 = 0; k0 < 256; k0 += 32) {
    v16bf a = load_a_tile(Ab, 256, mbase, k0, lane);
    accA = wmma_bf16(a, load_b_tile(Wg, 256, k0, nbase, lane), accA);
    accD = wmma_bf16(a, load_b_tile(Wg, 256, k0, 128 + nbase, lane), accD);
  }
  int g = lane >> 4, nl = lane & 15, c = nbase + nl;
  size_t rowbase = (size_t)b * NQ + mbase + 8 * g;
  const float* pxa = xa + rowbase * C_ + c;
  const float* pxd = xd + rowbase * C_ + c;
  float* po = out + ((size_t)b * C_ + c) * NQ + mbase + 8 * g;
#pragma unroll
  for (int r = 0; r < 8; ++r) {
    float ga = 1.0f / (1.0f + __expf(-accA[r]));
    float gd = 1.0f / (1.0f + __expf(-accD[r]));
    po[r] = ga * pxa[(size_t)r * C_] + gd * pxd[(size_t)r * C_];   // (B,C,H,W)
  }
}

extern "C" void kernel_launch(void* const* d_in, const int* in_sizes, int n_in,
                              void* d_out, int out_size, void* d_ws, size_t ws_size,
                              hipStream_t stream) {
  (void)in_sizes; (void)n_in; (void)out_size; (void)ws_size;
  const float* erp_f  = (const float*)d_in[0];
  const float* ico    = (const float*)d_in[1];
  const float* icoc   = (const float*)d_in[2];
  const float* Wq_a   = (const float*)d_in[3];
  const float* Wkv_a  = (const float*)d_in[4];
  const float* Wp_a   = (const float*)d_in[5];
  const float* bp_a   = (const float*)d_in[6];
  const float* Wq_d   = (const float*)d_in[7];
  const float* Wkv_d  = (const float*)d_in[8];
  const float* Wp_d   = (const float*)d_in[9];
  const float* bp_d   = (const float*)d_in[10];
  const float* Wdelta = (const float*)d_in[11];
  const float* Wga    = (const float*)d_in[12];
  const float* Wgd    = (const float*)d_in[13];

  char* w = (char*)d_ws;
  size_t off = 0;
  auto carve = [&](size_t bytes) -> void* {
    void* p = w + off;
    off = (off + bytes + 255) & ~(size_t)255;
    return p;
  };
  bf16_t* erp_bf  = (bf16_t*)carve((size_t)B_ * NQ * C_ * 2);
  bf16_t* ico_bf  = (bf16_t*)carve((size_t)B_ * NKP * C_ * 2);
  bf16_t* Wq_cat  = (bf16_t*)carve(128 * 256 * 2);
  bf16_t* Wkv_cat = (bf16_t*)carve(128 * 512 * 2);
  bf16_t* Wpa_bf  = (bf16_t*)carve(128 * 128 * 2);
  bf16_t* Wpd_bf  = (bf16_t*)carve(128 * 128 * 2);
  bf16_t* Wg_cat  = (bf16_t*)carve(256 * 256 * 2);
  float*  qcoord  = (float*)carve(NQ * 3 * 4);
  float*  wsum    = (float*)carve(3 * 4);
  bf16_t* q_a     = (bf16_t*)carve((size_t)B_ * NQ * C_ * 2);
  float*  q_d     = (float*)carve((size_t)B_ * NQ * C_ * 4);
  bf16_t* kTa     = (bf16_t*)carve((size_t)B_ * C_ * NKP * 2);
  bf16_t* va      = (bf16_t*)carve((size_t)B_ * NKP * C_ * 2);
  float*  kd      = (float*)carve((size_t)B_ * NKP * C_ * 4);
  bf16_t* vd      = (bf16_t*)carve((size_t)B_ * NKP * C_ * 2);
  float*  Sa      = (float*)carve((size_t)B_ * NQ * NKP * 4);
  float*  Sd      = (float*)carve((size_t)B_ * NQ * NKP * 4);
  bf16_t* Pa      = (bf16_t*)carve((size_t)B_ * NQ * NKP * 2);
  bf16_t* Pd      = (bf16_t*)carve((size_t)B_ * NQ * NKP * 2);
  float*  Oa      = (float*)carve((size_t)B_ * NQ * C_ * 4);
  float*  Od      = (float*)carve((size_t)B_ * NQ * C_ * 4);
  bf16_t* Xa      = (bf16_t*)carve((size_t)B_ * NQ * C_ * 2);
  bf16_t* Xd      = (bf16_t*)carve((size_t)B_ * NQ * C_ * 2);
  float*  xaf     = (float*)carve((size_t)B_ * NQ * C_ * 4);
  float*  xdf     = (float*)carve((size_t)B_ * NQ * C_ * 4);
  bf16_t* catb    = (bf16_t*)carve((size_t)B_ * NQ * 256 * 2);

  k_prep<<<3200, 256, 0, stream>>>(erp_f, ico, Wq_a, Wq_d, Wkv_a, Wkv_d, Wp_a, Wp_d,
                                   Wga, Wgd, Wdelta, erp_bf, ico_bf, Wq_cat, Wkv_cat,
                                   Wpa_bf, Wpd_bf, Wg_cat, qcoord, wsum);
  k_gemm_q<<<dim3(32, 16, B_), 32, 0, stream>>>(erp_bf, Wq_cat, q_a, q_d);
  k_gemm_kv<<<dim3(42, 32, B_), 32, 0, stream>>>(ico_bf, Wkv_cat, kTa, va, kd, vd);
  // affinity logits: S_a = q_a @ k^T   (M=512, N=672, K=128)
  k_gemm_f32out<<<dim3(32, 42, B_), 32, 0, stream>>>(q_a, C_, (long long)NQ * C_,
                                                     kTa, NKP, (long long)C_ * NKP,
                                                     Sa, NKP, (long long)NQ * NKP, C_);
  k_sdiff<<<dim3(NQ / QB, B_), 256, 0, stream>>>(q_d, qcoord, kd, icoc, wsum, Sd);
  k_softmax<<<B_ * NQ, 32, 0, stream>>>(Sa, Pa, 1.0f);
  k_softmax<<<B_ * NQ, 32, 0, stream>>>(Sd, Pd, 1.0f / 128.0f);
  // attn @ V   (M=512, N=128, K=672)
  k_gemm_f32out<<<dim3(32, 8, B_), 32, 0, stream>>>(Pa, NKP, (long long)NQ * NKP,
                                                    va, C_, (long long)NKP * C_,
                                                    Oa, C_, (long long)NQ * C_, NKP);
  k_gemm_f32out<<<dim3(32, 8, B_), 32, 0, stream>>>(Pd, NKP, (long long)NQ * NKP,
                                                    vd, C_, (long long)NKP * C_,
                                                    Od, C_, (long long)NQ * C_, NKP);
  k_perm_cvt<<<(B_ * NQ * C_ + 255) / 256, 256, 0, stream>>>(Oa, Od, Xa, Xd);
  k_gemm_p<<<dim3(32, 8, B_), 32, 0, stream>>>(Xa, Wpa_bf, bp_a, xaf, catb, 0);
  k_gemm_p<<<dim3(32, 8, B_), 32, 0, stream>>>(Xd, Wpd_bf, bp_d, xdf, catb, 128);
  k_gate_fuse<<<dim3(32, 8, B_), 32, 0, stream>>>(catb, Wg_cat, xaf, xdf, (float*)d_out);
}
